// ROPETransformerBlock_62569083568802
// MI455X (gfx1250) — matrix operand order repaired and checked by reading, not compiled, not run on hardware
//
#include <hip/hip_runtime.h>
#include <math.h>

typedef __bf16 bf16_t;
typedef __attribute__((ext_vector_type(16))) __bf16 v16bf;
typedef __attribute__((ext_vector_type(8)))  float  v8f;
typedef __attribute__((ext_vector_type(4)))  int    v4i;

union Frag {
    uint4  u[2];
    bf16_t h[16];
    v16bf  v;
};

static __device__ __forceinline__ v8f wmma_bf16(const Frag& a, const Frag& b, v8f c) {
    return __builtin_amdgcn_wmma_f32_16x16x32_bf16(
        /*neg_a=*/false, a.v, /*neg_b=*/false, b.v,
        /*c_mod=*/(short)0, c, /*reuse_a=*/false, /*reuse_b=*/false);
}

// --------------------------------------------------------------------------
// Async global->LDS (CDNA5 GLOBAL_LOAD_ASYNC_TO_LDS_B128, ASYNCcnt) with a
// synchronous fallback if this toolchain doesn't declare the builtin.
// Toolchain probe: builtin exists, signature (int4 as1*, int4 as3*, Ii, Ii).
// --------------------------------------------------------------------------
#if defined(__has_builtin)
#  if __has_builtin(__builtin_amdgcn_global_load_async_to_lds_b128)
#    define HAVE_ASYNC_LDS 1
#  endif
#endif
#ifndef HAVE_ASYNC_LDS
#  define HAVE_ASYNC_LDS 0
#endif

static __device__ __forceinline__ void copy16_g2lds(const bf16_t* g, bf16_t* l) {
#if HAVE_ASYNC_LDS
    __builtin_amdgcn_global_load_async_to_lds_b128(
        (__attribute__((address_space(1))) v4i*)(void*)g,
        (__attribute__((address_space(3))) v4i*)(void*)l, 0, 0);
#else
    *(uint4*)l = *(const uint4*)g;
#endif
}
// Wait until at most the 6 just-issued copies remain in flight (i.e. the
// previous stage's 6 copies have landed in LDS).
static __device__ __forceinline__ void async_wait_prev() {
#if HAVE_ASYNC_LDS
    asm volatile("s_wait_asynccnt 6" ::: "memory");
#endif
}
static __device__ __forceinline__ void async_wait_all() {
#if HAVE_ASYNC_LDS
    asm volatile("s_wait_asynccnt 0" ::: "memory");
#endif
}

// ---------------------------------------------------------------------------
// Weight transpose + fp32 -> bf16 convert.  W:[R,C] row-major  ->  Wt:[C,R]
// ---------------------------------------------------------------------------
__global__ void transpose_to_bf16(const float* __restrict__ W,
                                  bf16_t* __restrict__ Wt,
                                  int R, int C) {
    size_t idx = (size_t)blockIdx.x * blockDim.x + threadIdx.x;
    if (idx >= (size_t)R * C) return;
    size_t c = idx / R;          // Wt row  (GEMM k)
    size_t r = idx % R;          // Wt col  (GEMM n)
    Wt[idx] = (bf16_t)W[r * (size_t)C + c];
}

// ---------------------------------------------------------------------------
// LayerNorm (fp32 in) -> bf16 out.  One block (256 thr) per row of D elems.
// ---------------------------------------------------------------------------
__global__ __launch_bounds__(256)
void layernorm_to_bf16(const float* __restrict__ x,
                       const float* __restrict__ g,
                       const float* __restrict__ be,
                       bf16_t* __restrict__ out, int D) {
    __shared__ float red[256];
    const size_t row = blockIdx.x;
    const float* xr = x + row * (size_t)D;
    const int t = threadIdx.x;

    float s = 0.f;
    for (int i = t; i < D; i += 256) s += xr[i];
    red[t] = s; __syncthreads();
    for (int st = 128; st > 0; st >>= 1) {
        if (t < st) red[t] += red[t + st];
        __syncthreads();
    }
    const float mu = red[0] / (float)D;
    __syncthreads();

    float v = 0.f;
    for (int i = t; i < D; i += 256) { float d = xr[i] - mu; v += d * d; }
    red[t] = v; __syncthreads();
    for (int st = 128; st > 0; st >>= 1) {
        if (t < st) red[t] += red[t + st];
        __syncthreads();
    }
    const float rstd = rsqrtf(red[0] / (float)D + 1e-5f);

    for (int i = t; i < D; i += 256)
        out[row * (size_t)D + i] = (bf16_t)((xr[i] - mu) * rstd * g[i] + be[i]);
}

// ---------------------------------------------------------------------------
// RoPE in place on Q [B*T, D]; head dim = 64, theta = 10000.
// ---------------------------------------------------------------------------
__global__ void rope_q(bf16_t* __restrict__ X, int T, int D) {
    size_t pidx  = (size_t)blockIdx.x * blockDim.x + threadIdx.x;
    size_t halfD = (size_t)D / 2;
    size_t row = pidx / halfD;
    int    cp  = (int)(pidx % halfD);
    int    tpos = (int)(row % (size_t)T);
    int    dd   = (cp * 2) & 63;
    float  ang  = (float)tpos * __powf(10000.f, -(float)dd / 64.f);
    float  c = __cosf(ang), s = __sinf(ang);
    bf16_t* p = X + row * (size_t)D + (size_t)cp * 2;
    float xe = (float)p[0], xo = (float)p[1];
    p[0] = (bf16_t)(xe * c - xo * s);
    p[1] = (bf16_t)(xo * c + xe * s);
}

// ---------------------------------------------------------------------------
// RoPE on K + repack into transposed layout Kt[B, H, 64, T] so that the
// attention B-fragment (lane = key column, elems = head-dim) loads coalesce across lanes.
// ---------------------------------------------------------------------------
__global__ void rope_k_transpose(const bf16_t* __restrict__ K,
                                 bf16_t* __restrict__ Kt,
                                 int T, int D, int H) {
    size_t pidx  = (size_t)blockIdx.x * blockDim.x + threadIdx.x;
    size_t halfD = (size_t)D / 2;
    size_t row = pidx / halfD;           // b*T + t
    int    cp  = (int)(pidx % halfD);
    int    tpos = (int)(row % (size_t)T);
    int    b    = (int)(row / (size_t)T);
    int    col  = cp * 2;
    int    head = col >> 6;
    int    dd   = col & 63;
    float  ang  = (float)tpos * __powf(10000.f, -(float)dd / 64.f);
    float  c = __cosf(ang), s = __sinf(ang);
    const bf16_t* p = K + row * (size_t)D + col;
    float xe = (float)p[0], xo = (float)p[1];
    size_t base = ((size_t)(b * H + head) * 64) * (size_t)T;
    Kt[base + (size_t)dd       * T + tpos] = (bf16_t)(xe * c - xo * s);
    Kt[base + (size_t)(dd + 1) * T + tpos] = (bf16_t)(xo * c + xe * s);
}

// ---------------------------------------------------------------------------
// Tiled bf16 WMMA GEMM:  C[M,N] = A[M,K] @ Bt[K,N] + bias  (+ epilogue)
//   EPI 0: out = bf16(acc + bias)
//   EPI 1: out = bf16(gelu(acc + bias))          (exact erf GELU)
//   EPI 2: out = f32(res + acc + bias)           (residual add)
// Block: 256 thr (8 waves), tile 128x64, BK=64, double-buffered LDS with
// ASYNCcnt software pipelining: the next slab's async copies are issued
// before computing on the current slab; s_wait_asynccnt 6 releases only the
// previous stage's copies.  Each wave owns a 2x2 grid of 16x16 WMMA tiles
// (8 WMMAs per slab reusing 2 A-frags x 2 B-frags per 32-wide k-chunk).
// ---------------------------------------------------------------------------
#define BM 128
#define BN 64
#define BK 64

template <int EPI>
__global__ __launch_bounds__(256)
void gemm_bf16_wmma(const bf16_t* __restrict__ A,
                    const bf16_t* __restrict__ Bt,
                    const float*  __restrict__ bias,
                    const float*  __restrict__ res,
                    void*         __restrict__ out,
                    int M, int N, int K) {
    __shared__ bf16_t As[2][BM][BK];   // 2 x 16 KB
    __shared__ bf16_t Bs[2][BK][BN];   // 2 x  8 KB

    const int m0   = blockIdx.y * BM;
    const int n0   = blockIdx.x * BN;
    const int tid  = threadIdx.x;
    const int wave = tid >> 5;
    const int lane = tid & 31;
    const int half = lane >> 4;
    const int l16  = lane & 15;
    const int wm   = (wave >> 1) * 2;     // first of 2 m-tiles (0,2,4,6)
    const int wn   = (wave & 1) * 2;      // first of 2 n-tiles (0,2)

    // Stage one K-slab into LDS buffer `buf`: 6 x 16 B async copies/thread.
    auto stage = [&](int buf, int k0) {
#pragma unroll
        for (int ch = 0; ch < 4; ++ch) {        // A: 128x64 bf16 = 16 KB
            const int c   = tid + ch * 256;     // 0..1023
            const int r   = c >> 3;
            const int col = (c & 7) * 8;
            copy16_g2lds(A + (size_t)(m0 + r) * K + k0 + col, &As[buf][r][col]);
        }
#pragma unroll
        for (int ch = 0; ch < 2; ++ch) {        // B: 64x64 bf16 = 8 KB
            const int c   = tid + ch * 256;     // 0..511
            const int r   = c >> 3;
            const int col = (c & 7) * 8;
            copy16_g2lds(Bt + (size_t)(k0 + r) * N + n0 + col, &Bs[buf][r][col]);
        }
    };

    v8f acc[2][2] = {{v8f{}, v8f{}}, {v8f{}, v8f{}}};

    const int nSlab = K / BK;
    stage(0, 0);
    for (int s = 0; s < nSlab; ++s) {
        const int cur = s & 1;
        if (s + 1 < nSlab) {
            stage(cur ^ 1, (s + 1) * BK);       // issue next slab (in flight)
            async_wait_prev();                  // current slab's copies landed
        } else {
            async_wait_all();
        }
        if (s + 2 < nSlab) {                    // global_prefetch_b8, 2 ahead
            const int kp = (s + 2) * BK;
            __builtin_prefetch(A  + (size_t)(m0 + (tid >> 1)) * K + kp, 0, 3);
            __builtin_prefetch(Bt + (size_t)(kp + (tid >> 2)) * N + n0, 0, 3);
        }
        __syncthreads();

#pragma unroll
        for (int kc = 0; kc < 2; ++kc) {
            // A fragments per ISA 16x32 layout: lane = M, half splits K.
            Frag fa[2], fb[2];
#pragma unroll
            for (int i = 0; i < 2; ++i) {
                const int ar = (wm + i) * 16 + l16;
                fa[i].u[0] = *(const uint4*)(&As[cur][ar][kc * 32 + half * 8]);
                fa[i].u[1] = *(const uint4*)(&As[cur][ar][kc * 32 + 16 + half * 8]);
            }
            // B fragments: lane = n (l16), 16 k values in table order (same as A).
#pragma unroll
            for (int j = 0; j < 2; ++j) {
#pragma unroll
                for (int s = 0; s < 16; ++s)
                    fb[j].h[s] = Bs[cur][kc * 32 + (s < 8 ? half * 8 + s : 16 + half * 8 + (s - 8))][(wn + j) * 16 + l16];
            }
#pragma unroll
            for (int i = 0; i < 2; ++i)
#pragma unroll
                for (int j = 0; j < 2; ++j)
                    acc[i][j] = wmma_bf16(fa[i], fb[j], acc[i][j]);
        }
        __syncthreads();   // all waves done reading before buffer reuse
    }

    // Epilogue. C layout: lane -> n = l16, VGPR v -> m = v + 8*half.
#pragma unroll
    for (int i = 0; i < 2; ++i) {
        const int mBase = m0 + (wm + i) * 16 + half * 8;
#pragma unroll
        for (int j = 0; j < 2; ++j) {
            const int n  = n0 + (wn + j) * 16 + l16;
            const float bn = bias[n];
#pragma unroll
            for (int v = 0; v < 8; ++v) {
                const size_t ofs = (size_t)(mBase + v) * N + n;
                float val = acc[i][j][v] + bn;
                if (EPI == 0) {
                    ((bf16_t*)out)[ofs] = (bf16_t)val;
                } else if (EPI == 1) {
                    float ge = 0.5f * val * (1.0f + erff(val * 0.70710678118f));
                    ((bf16_t*)out)[ofs] = (bf16_t)ge;
                } else {
                    ((float*)out)[ofs] = res[ofs] + val;
                }
            }
        }
    }
}

// ---------------------------------------------------------------------------
// Causal flash attention, bf16 in/out, fp32 online softmax.
// Q/V/O layout: [B*T, 1024] heads inline; K pre-transposed as Kt[B,H,64,T].
// One wave per (16-row Q tile, head, batch).  Q fragments are
// contiguous 2x16 B loads, K/V gathered element-wise; only P round-trips through LDS (layout change).
// ---------------------------------------------------------------------------
__global__ __launch_bounds__(32)
void attn_flash_wmma(const bf16_t* __restrict__ Q,
                     const bf16_t* __restrict__ Kt,
                     const bf16_t* __restrict__ Vb,
                     bf16_t* __restrict__ Ob,
                     int T, int D, int H) {
    __shared__ bf16_t Ps[16][32];   // P tile: 16 rows x 32 cols (1 KB)

    const int t0   = blockIdx.x * 16;
    const int h    = blockIdx.y;
    const int b    = blockIdx.z;
    const int lane = threadIdx.x;
    const int half = lane >> 4;
    const int l16  = lane & 15;
    const size_t baseRow = (size_t)b * T;
    const int hoff = h * 64;
    const bf16_t* KtB = Kt + ((size_t)(b * H + h) * 64) * (size_t)T;

    // Q A-fragments for the two 32-wide K-dim chunks (head dim = 64).
    Frag fq[2];
    {
        const bf16_t* qrow = Q + (baseRow + t0 + l16) * (size_t)D + hoff;
#pragma unroll
        for (int c = 0; c < 2; ++c) {
            fq[c].u[0] = *(const uint4*)(qrow + c * 32 + half * 8);
            fq[c].u[1] = *(const uint4*)(qrow + c * 32 + 16 + half * 8);
        }
    }

    float rm[8], rl[8];
    v8f O[4] = {v8f{}, v8f{}, v8f{}, v8f{}};
#pragma unroll
    for (int v = 0; v < 8; ++v) { rm[v] = -3.0e38f; rl[v] = 0.f; }

    const int ssEnd = t0 + 16;     // causal: keys with ss <= ts only
    for (int ss0 = 0; ss0 < ssEnd; ss0 += 32) {
        // S tiles (two 16-wide column sub-tiles); K-dim = 64 -> 2 WMMAs each.
        // B-fragment: lane = key column (l16), 16 kd values in table order from Kt.
        v8f S[2] = {v8f{}, v8f{}};
#pragma unroll
        for (int kc = 0; kc < 2; ++kc) {
            const bf16_t* krow = KtB + ss0 + l16;
#pragma unroll
            for (int sub = 0; sub < 2; ++sub) {
                Frag fk;
#pragma unroll
                for (int s = 0; s < 16; ++s)
                    fk.h[s] = krow[(size_t)(kc * 32 + (s < 8 ? half * 8 + s : 16 + half * 8 + (s - 8))) * T + sub * 16];
                S[sub] = wmma_bf16(fq[kc], fk, S[sub]);
            }
        }

        // Scale, causal mask, online softmax update.
        float Sv[2][8];
        float alpha[8];
#pragma unroll
        for (int v = 0; v < 8; ++v) {
            const int ts = t0 + v + 8 * half;
            float mx = -3.0e38f;
#pragma unroll
            for (int sub = 0; sub < 2; ++sub) {
                const int ss = ss0 + sub * 16 + l16;
                float s = S[sub][v] * 0.125f;      // HEAD_DIM^-0.5
                if (ss > ts) s = -3.0e38f;
                Sv[sub][v] = s;
                mx = fmaxf(mx, s);
            }
#pragma unroll
            for (int off = 1; off < 16; off <<= 1)
                mx = fmaxf(mx, __shfl_xor(mx, off, 32));
            const float mnew = fmaxf(rm[v], mx);
            alpha[v] = __expf(rm[v] - mnew);
            rm[v] = mnew;
            float rowsum = 0.f;
#pragma unroll
            for (int sub = 0; sub < 2; ++sub) {
                float pv = __expf(Sv[sub][v] - mnew);
                Sv[sub][v] = pv;
                rowsum += pv;
            }
#pragma unroll
            for (int off = 1; off < 16; off <<= 1)
                rowsum += __shfl_xor(rowsum, off, 32);
            rl[v] = rl[v] * alpha[v] + rowsum;
#pragma unroll
            for (int d = 0; d < 4; ++d) O[d][v] *= alpha[v];
        }

        // P: C-layout -> LDS -> A-layout fragment.
        __syncthreads();
#pragma unroll
        for (int sub = 0; sub < 2; ++sub)
#pragma unroll
            for (int v = 0; v < 8; ++v)
                Ps[v + 8 * half][sub * 16 + l16] = (bf16_t)Sv[sub][v];
        __syncthreads();

        Frag fp;
        fp.u[0] = *(const uint4*)(&Ps[l16][half * 8]);
        fp.u[1] = *(const uint4*)(&Ps[l16][16 + half * 8]);

        // O += P (16x32) x V (32x64): 4 d-tiles, V B-fragment straight from
        // global (lane = dim l16, 16 key rows in table order).
        const bf16_t* vrow = Vb + baseRow * (size_t)D + hoff + l16;
#pragma unroll
        for (int d = 0; d < 4; ++d) {
            Frag fv;
#pragma unroll
            for (int s = 0; s < 16; ++s) {
                int vr = ss0 + (s < 8 ? half * 8 + s : 16 + half * 8 + (s - 8)); if (vr > T - 1) vr = T - 1;
                fv.h[s] = vrow[(size_t)vr * D + d * 16];
            }
            O[d] = wmma_bf16(fp, fv, O[d]);
        }
    }

    // Normalize and store bf16 (heads-inline layout feeds the Wo GEMM).
#pragma unroll
    for (int d = 0; d < 4; ++d)
#pragma unroll
        for (int v = 0; v < 8; ++v) {
            const int m = t0 + v + 8 * half;
            Ob[(baseRow + m) * (size_t)D + hoff + d * 16 + l16] =
                (bf16_t)(O[d][v] / rl[v]);
        }
}

// ---------------------------------------------------------------------------
// Host launcher
// ---------------------------------------------------------------------------
extern "C" void kernel_launch(void* const* d_in, const int* in_sizes, int n_in,
                              void* d_out, int out_size, void* d_ws, size_t ws_size,
                              hipStream_t stream) {
    (void)in_sizes; (void)n_in; (void)out_size; (void)ws_size;

    constexpr int Bb = 2, T = 2048, D = 1024, H = 16, F = 4096;
    constexpr int M = Bb * T;                       // 4096 rows

    const float* src = (const float*)d_in[0];
    // d_in[1] = attn_mask (causal triu) — structure known, not read.
    const float* Wq = (const float*)d_in[2];  const float* bq = (const float*)d_in[3];
    const float* Wk = (const float*)d_in[4];  const float* bk = (const float*)d_in[5];
    const float* Wv = (const float*)d_in[6];  const float* bv = (const float*)d_in[7];
    const float* Wo = (const float*)d_in[8];  const float* bo = (const float*)d_in[9];
    const float* W1 = (const float*)d_in[10]; const float* b1 = (const float*)d_in[11];
    const float* W2 = (const float*)d_in[12]; const float* b2 = (const float*)d_in[13];
    const float* g1 = (const float*)d_in[14]; const float* be1 = (const float*)d_in[15];
    const float* g2 = (const float*)d_in[16]; const float* be2 = (const float*)d_in[17];

    char* p = (char*)d_ws;
    auto carve = [&](size_t bytes) -> char* {
        char* r = p; p += (bytes + 255) & ~(size_t)255; return r;
    };
    bf16_t* xln1 = (bf16_t*)carve((size_t)M * D * 2);
    bf16_t* xln2 = (bf16_t*)carve((size_t)M * D * 2);
    bf16_t* Wqt  = (bf16_t*)carve((size_t)D * D * 2);
    bf16_t* Wkt  = (bf16_t*)carve((size_t)D * D * 2);
    bf16_t* Wvt  = (bf16_t*)carve((size_t)D * D * 2);
    bf16_t* Wot  = (bf16_t*)carve((size_t)D * D * 2);
    bf16_t* W1t  = (bf16_t*)carve((size_t)D * F * 2);
    bf16_t* W2t  = (bf16_t*)carve((size_t)F * D * 2);
    bf16_t* Qb   = (bf16_t*)carve((size_t)M * D * 2);
    bf16_t* Kb   = (bf16_t*)carve((size_t)M * D * 2);
    bf16_t* KtT  = (bf16_t*)carve((size_t)M * D * 2);   // [B,H,64,T]
    bf16_t* Vb   = (bf16_t*)carve((size_t)M * D * 2);
    bf16_t* attn = (bf16_t*)carve((size_t)M * D * 2);
    float*  src2 = (float*) carve((size_t)M * D * 4);
    bf16_t* hbuf = (bf16_t*)carve((size_t)M * F * 2);

    // 1) Weight transposes to bf16 [K,N].
    const int nDD = (D * D + 255) / 256;
    transpose_to_bf16<<<nDD, 256, 0, stream>>>(Wq, Wqt, D, D);
    transpose_to_bf16<<<nDD, 256, 0, stream>>>(Wk, Wkt, D, D);
    transpose_to_bf16<<<nDD, 256, 0, stream>>>(Wv, Wvt, D, D);
    transpose_to_bf16<<<nDD, 256, 0, stream>>>(Wo, Wot, D, D);
    const int nDF = (D * F + 255) / 256;
    transpose_to_bf16<<<nDF, 256, 0, stream>>>(W1, W1t, F, D);   // [F,D]->[D,F]
    transpose_to_bf16<<<nDF, 256, 0, stream>>>(W2, W2t, D, F);   // [D,F]->[F,D]

    // 2) LN1.
    layernorm_to_bf16<<<M, 256, 0, stream>>>(src, g1, be1, xln1, D);

    // 3) QKV projections.
    dim3 gDD(D / BN, M / BM);
    gemm_bf16_wmma<0><<<gDD, 256, 0, stream>>>(xln1, Wqt, bq, nullptr, Qb, M, D, D);
    gemm_bf16_wmma<0><<<gDD, 256, 0, stream>>>(xln1, Wkt, bk, nullptr, Kb, M, D, D);
    gemm_bf16_wmma<0><<<gDD, 256, 0, stream>>>(xln1, Wvt, bv, nullptr, Vb, M, D, D);

    // 4) RoPE on Q (in place) and K (fused with transpose to Kt[B,H,64,T]).
    const size_t pairs = (size_t)M * D / 2;
    const int nRope = (int)((pairs + 255) / 256);
    rope_q<<<nRope, 256, 0, stream>>>(Qb, T, D);
    rope_k_transpose<<<nRope, 256, 0, stream>>>(Kb, KtT, T, D, H);

    // 5) Causal flash attention.
    attn_flash_wmma<<<dim3(T / 16, H, Bb), 32, 0, stream>>>(Qb, KtT, Vb, attn, T, D, H);

    // 6) Output projection + residual -> src2 (f32).
    gemm_bf16_wmma<2><<<gDD, 256, 0, stream>>>(attn, Wot, bo, src, src2, M, D, D);

    // 7) LN2.
    layernorm_to_bf16<<<M, 256, 0, stream>>>(src2, g2, be2, xln2, D);

    // 8) FFN1 + exact GELU.
    dim3 gDF(F / BN, M / BM);
    gemm_bf16_wmma<1><<<gDF, 256, 0, stream>>>(xln2, W1t, b1, nullptr, hbuf, M, F, D);

    // 9) FFN2 + residual -> d_out (f32).
    gemm_bf16_wmma<2><<<gDD, 256, 0, stream>>>(hbuf, W2t, b2, src2, (float*)d_out, M, D, F);
}
